// Tacotron2_61838939127911
// MI455X (gfx1250) — matrix operand order repaired and checked by reading, not compiled, not run on hardware
//
#include <hip/hip_runtime.h>
#include <hip/hip_bf16.h>
#include <math.h>

// ---------------- Problem constants ----------------
#define B_SZ      64
#define T_ENC     512
#define T_DEC     800
#define N_MEL     80
#define ENC_DIM   512
#define ATTN_RNN  1024
#define DEC_RNN   1024
#define PRENET    256
#define ATTN_DIM  128
#define N_FILT    32
#define KSIZE     31

// Combined LSTM K dims
#define K_ATTN    1792   // 768 (x,ctx) + 1024 (h)   -> 56 kTiles of 32
#define K_DEC     2560   // 1536 (h,ctx) + 1024 (h)  -> 80 kTiles of 32
#define KT_ATTN   56
#define KT_DEC    80
#define NT_GATES  256    // 4096 / 16

typedef __bf16 v16bf __attribute__((ext_vector_type(16)));
typedef float  v8f   __attribute__((ext_vector_type(8)));

__device__ __forceinline__ unsigned short f2bf(float x) {
  union { float f; unsigned u; } v; v.f = x;
  unsigned r = v.u + 0x7FFFu + ((v.u >> 16) & 1u);
  return (unsigned short)(r >> 16);
}
__device__ __forceinline__ float bf2f(unsigned short u) {
  union { unsigned u32; float f; } v; v.u32 = ((unsigned)u) << 16; return v.f;
}
__device__ __forceinline__ float sigm(float x) { return 1.f / (1.f + __expf(-x)); }

__device__ __forceinline__ v8f wmma_bf16(v16bf a, v16bf b, v8f c) {
  return __builtin_amdgcn_wmma_f32_16x16x32_bf16(false, a, false, b, (short)0, c, false, false);
}

// ---------------------------------------------------------------------------
// Pack an output-major fp32 weight [N x K] (optionally two concatenated
// sources along K) into WMMA B-fragment order:
//   dst[(kt*NT + nt)*512 + lane*16 + j], k = kt*32 + (j&7) + 8*(lane>=16) + 16*(j>=8),
//   n = nt*16 + (lane&15).   Zero-pads k in [K0+K1, Kpad).
// ---------------------------------------------------------------------------
__global__ void pack_bfrag(unsigned short* __restrict__ dst,
                           const float* __restrict__ src0, int ld0, int K0,
                           const float* __restrict__ src1, int ld1,
                           int Kpad, int NT) {
  long id = (long)blockIdx.x * 256 + threadIdx.x;
  long total = (long)(Kpad >> 5) * NT * 512;
  if (id >= total) return;
  int j = (int)(id & 15);
  int lane = (int)((id >> 4) & 31);
  long t = id >> 9;
  int nt = (int)(t % NT);
  int kt = (int)(t / NT);
  int n = nt * 16 + (lane & 15);
  int k = kt * 32 + (j & 7) + 8 * (lane >> 4) + 16 * (j >> 3);
  float v = 0.f;
  if (k < K0)                        v = src0[(long)n * ld0 + k];
  else if (src1)                     v = src1[(long)n * ld1 + (k - K0)];
  dst[id] = f2bf(v);
}

// ---------------------------------------------------------------------------
// Pack per-step activations [64 x K] into WMMA A-fragment order:
//   dst[(kt*4 + mt)*512 + lane*16 + j]
//   m = mt*16 + (lane&15),  k = kt*32 + (j&7) + 8*(lane>=16) + 16*(j>=8)
// Sources: optional bf16 segment then up to three fp32 segments (row strides
// equal segment length).
// ---------------------------------------------------------------------------
__global__ void packA(unsigned short* __restrict__ dst, int KT,
                      const unsigned short* __restrict__ bseg, int blen,
                      const float* __restrict__ f1, int l1,
                      const float* __restrict__ f2, int l2,
                      const float* __restrict__ f3, int l3) {
  int id = blockIdx.x * 256 + threadIdx.x;
  int total = KT * 4 * 512;
  if (id >= total) return;
  int j = id & 15;
  int lane = (id >> 4) & 31;
  int t = id >> 9;
  int mt = t & 3;
  int kt = t >> 2;
  int m = mt * 16 + (lane & 15);
  int k = kt * 32 + (j & 7) + 8 * (lane >> 4) + 16 * (j >> 3);
  unsigned short out;
  if (k < blen) {
    out = bseg[m * blen + k];
  } else {
    int kk = k - blen;
    float v;
    if (kk < l1)             v = f1[m * l1 + kk];
    else if ((kk -= l1) < l2) v = f2[m * l2 + kk];
    else                      v = f3[m * l3 + (kk - l2)];
    out = f2bf(v);
  }
  dst[id] = out;
}

// ---------------------------------------------------------------------------
// Combined LSTM gate GEMM: gates[64 x 4096] = A[64 x K] * Wcat^T + bias.
// Grid: 4 mTiles * 32 nGroups = 128 blocks of 8 waves; wave -> one 16x16 tile.
// A fragments (shared by all waves of the block) are staged into LDS with
// CDNA5 async global->LDS copies, then the k-loop is pure WMMA.
// ---------------------------------------------------------------------------
__global__ void __launch_bounds__(256) lstm_gemm(
    const unsigned short* __restrict__ Apack,   // [(kt*4+mt)*512 + ...]
    const unsigned short* __restrict__ Bpack,   // [(kt*256+nt)*512 + ...]
    const float* __restrict__ bias,             // [4096]
    float* __restrict__ gates,                  // [64 x 4096]
    int KT) {
  __shared__ __align__(64) unsigned short ldsA[KT_DEC * 512];  // 80 KB max
  const int tid = threadIdx.x;
  const int lane = tid & 31;
  const int wave = tid >> 5;
  const int mt = blockIdx.x >> 5;
  const int ng = blockIdx.x & 31;
  const int nTile = ng * 8 + wave;

  // Stage this mTile's A fragments: KT * 1KB, 16B chunks, async to LDS.
  {
    unsigned lbase = (unsigned)(size_t)(&ldsA[0]);   // LDS offset (low 32 bits)
    int nchunks = KT * 64;
    for (int c = tid; c < nchunks; c += 256) {
      int kt = c >> 6, w = c & 63;
      unsigned loff = lbase + (unsigned)(kt * 1024 + w * 16);
      unsigned goff = (unsigned)((kt * 4 + mt) * 1024 + w * 16);
      asm volatile("global_load_async_to_lds_b128 %0, %1, %2"
                   :: "v"(loff), "v"(goff), "s"(Apack) : "memory");
    }
    asm volatile("s_wait_asynccnt 0" ::: "memory");
  }
  __syncthreads();

  v8f acc = {};
  for (int kt = 0; kt < KT; ++kt) {
    v16bf a = *reinterpret_cast<const v16bf*>(&ldsA[kt * 512 + lane * 16]);
    v16bf b = *reinterpret_cast<const v16bf*>(
        &Bpack[((size_t)(kt * NT_GATES + nTile)) * 512 + lane * 16]);
    acc = wmma_bf16(a, b, acc);
  }

  const int n = nTile * 16 + (lane & 15);
  const int mbase = mt * 16 + 8 * (lane >> 4);
  const float bv = bias[n];
#pragma unroll
  for (int r = 0; r < 8; ++r)
    gates[(size_t)(mbase + r) * 4096 + n] = acc[r] + bv;
}

// LSTM pointwise: (i,f,g,o) -> h,c.  65536 threads.
__global__ void lstm_point(const float* __restrict__ g,
                           float* __restrict__ h, float* __restrict__ c) {
  int id = blockIdx.x * 256 + threadIdx.x;
  int m = id >> 10, u = id & 1023;
  const float* row = g + (size_t)m * 4096;
  float gi = row[u], gf = row[1024 + u], gg = row[2048 + u], go = row[3072 + u];
  float cc = sigm(gf) * c[id] + sigm(gi) * tanhf(gg);
  c[id] = cc;
  h[id] = sigm(go) * tanhf(cc);
}

// ---------------------------------------------------------------------------
// Fused location-sensitive attention: pq = ah@Wq^T, 1D conv over (aw,awc),
// energies + mask + softmax, ctx = aw_new @ memory, cumulative update.
// One block per batch element.
// ---------------------------------------------------------------------------
__global__ void __launch_bounds__(256) attn_kernel(
    const float* __restrict__ ah, const float* __restrict__ Wq,
    const float* __restrict__ pmem,
    float* __restrict__ aw, float* __restrict__ awc,
    const float* __restrict__ Wloc, const float* __restrict__ Wld,
    const float* __restrict__ vvec,
    const unsigned short* __restrict__ membf, const int* __restrict__ mlen,
    float* __restrict__ ctx, float* __restrict__ align_out, int s) {
  __shared__ float pq[128], aw_s[512], awc_s[512], e_s[512], red[256];
  __shared__ float wld_s[ATTN_DIM * N_FILT], wloc_s[N_FILT * 2 * KSIZE], v_s[128];
  const int b = blockIdx.x, tid = threadIdx.x;

  for (int i = tid; i < 512; i += 256) { aw_s[i] = aw[b * 512 + i]; awc_s[i] = awc[b * 512 + i]; }
  for (int i = tid; i < ATTN_DIM * N_FILT; i += 256) wld_s[i] = Wld[i];
  for (int i = tid; i < N_FILT * 2 * KSIZE; i += 256) wloc_s[i] = Wloc[i];
  if (tid < 128) {
    v_s[tid] = vvec[tid];
    const float* a = ah + (size_t)b * 1024;
    const float* w = Wq + (size_t)tid * 1024;
    float s0 = 0.f;
    for (int jj = 0; jj < 1024; ++jj) s0 += a[jj] * w[jj];
    pq[tid] = s0;
  }
  __syncthreads();

  const int len = mlen[b];
  for (int t = tid; t < 512; t += 256) {
    float e;
    if (t >= len) {
      e = -__builtin_inff();
    } else {
      float loc[N_FILT];
#pragma unroll
      for (int f = 0; f < N_FILT; ++f) loc[f] = 0.f;
      for (int kk = 0; kk < KSIZE; ++kk) {
        int tt = t + kk - (KSIZE - 1) / 2;
        float a0 = (tt >= 0 && tt < 512) ? aw_s[tt] : 0.f;
        float c0 = (tt >= 0 && tt < 512) ? awc_s[tt] : 0.f;
#pragma unroll
        for (int f = 0; f < N_FILT; ++f)
          loc[f] += wloc_s[f * 62 + kk] * a0 + wloc_s[f * 62 + KSIZE + kk] * c0;
      }
      const float* pm = pmem + ((size_t)b * 512 + t) * 128;
      e = 0.f;
      for (int d = 0; d < 128; ++d) {
        float s0 = pq[d] + pm[d];
#pragma unroll
        for (int f = 0; f < N_FILT; ++f) s0 += wld_s[d * 32 + f] * loc[f];
        e += v_s[d] * tanhf(s0);
      }
    }
    e_s[t] = e;
  }
  __syncthreads();

  // softmax over 512 entries
  red[tid] = fmaxf(e_s[tid], e_s[tid + 256]);
  __syncthreads();
  for (int o = 128; o > 0; o >>= 1) {
    if (tid < o) red[tid] = fmaxf(red[tid], red[tid + o]);
    __syncthreads();
  }
  float mx = red[0];
  __syncthreads();
  red[tid] = __expf(e_s[tid] - mx) + __expf(e_s[tid + 256] - mx);
  __syncthreads();
  for (int o = 128; o > 0; o >>= 1) {
    if (tid < o) red[tid] += red[tid + o];
    __syncthreads();
  }
  float inv = 1.f / red[0];
  __syncthreads();

  for (int t = tid; t < 512; t += 256) {
    float a0 = __expf(e_s[t] - mx) * inv;
    e_s[t] = a0;
    aw[b * 512 + t] = a0;
    awc[b * 512 + t] = awc_s[t] + a0;
    align_out[((size_t)b * T_DEC + s) * 512 + t] = a0;
  }
  __syncthreads();

  // ctx[b, d] = sum_t aw_new[t] * memory[b, t, d]  (bf16 memory copy, L2-hot)
  for (int d = tid; d < 512; d += 256) {
    const unsigned short* mcol = membf + (size_t)b * 512 * 512 + d;
    float acc = 0.f;
    for (int t = 0; t < 512; ++t) acc += e_s[t] * bf2f(mcol[(size_t)t * 512]);
    ctx[b * 512 + d] = acc;
  }
}

// mel / gate projections. One block per batch element, 128 threads.
__global__ void __launch_bounds__(128) proj_kernel(
    const float* __restrict__ dh, const float* __restrict__ ctx,
    const float* __restrict__ Wp, const float* __restrict__ bp,
    const float* __restrict__ Wg, const float* __restrict__ bg,
    float* __restrict__ mel_out, float* __restrict__ gate_out, int s) {
  __shared__ float hc[1536];
  const int b = blockIdx.x, tid = threadIdx.x;
  for (int i = tid; i < 1024; i += 128) hc[i] = dh[b * 1024 + i];
  for (int i = tid; i < 512; i += 128) hc[1024 + i] = ctx[b * 512 + i];
  __syncthreads();
  if (tid < N_MEL) {
    const float* w = Wp + (size_t)tid * 1536;
    float s0 = bp[tid];
    for (int j = 0; j < 1536; ++j) s0 += hc[j] * w[j];
    mel_out[((size_t)b * T_DEC + s) * N_MEL + tid] = s0;
  } else if (tid == N_MEL) {
    float s0 = bg[0];
    for (int j = 0; j < 1536; ++j) s0 += hc[j] * Wg[j];
    gate_out[(size_t)b * T_DEC + s] = s0;
  }
}

// pmem = memory @ Wm^T, plus bf16 copy of memory. One block per (b,t) row.
__global__ void __launch_bounds__(256) pmem_kernel(
    const float* __restrict__ memory, const float* __restrict__ Wm,
    float* __restrict__ pmem, unsigned short* __restrict__ membf) {
  __shared__ float row[512];
  const int bt = blockIdx.x;  // 0 .. 64*512-1
  const int tid = threadIdx.x;
  const float* src = memory + (size_t)bt * 512;
  for (int i = tid; i < 512; i += 256) {
    float x = src[i];
    row[i] = x;
    membf[(size_t)bt * 512 + i] = f2bf(x);
  }
  __syncthreads();
  if (tid < 128) {
    const float* w = Wm + (size_t)tid * 512;
    float s0 = 0.f;
    for (int j = 0; j < 512; ++j) s0 += row[j] * w[j];
    pmem[(size_t)bt * 128 + tid] = s0;
  }
}

// Zero recurrent state + combine biases.
__global__ void init_kernel(float* ah, float* ac, float* dh, float* dc,
                            float* aw, float* awc, float* ctx,
                            const float* bia, const float* bha, float* bias_a,
                            const float* bid, const float* bhd, float* bias_d) {
  int id = blockIdx.x * 256 + threadIdx.x;
  if (id < 65536) { ah[id] = 0.f; ac[id] = 0.f; dh[id] = 0.f; dc[id] = 0.f; }
  if (id < 32768) { aw[id] = 0.f; awc[id] = 0.f; ctx[id] = 0.f; }
  if (id < 4096)  { bias_a[id] = bia[id] + bha[id]; bias_d[id] = bid[id] + bhd[id]; }
}

// ---------------------------------------------------------------------------
// Fused prenet over all 51200 rows (row = s*64 + b), two WMMA stages.
// Stage-1 output is written directly into stage-2 A-fragment layout in LDS.
// ---------------------------------------------------------------------------
__global__ void __launch_bounds__(256) prenet_kernel(
    const float* __restrict__ dec_in,          // [T_DEC, 64, 80]
    const unsigned short* __restrict__ W1p,    // K=96 pad, N=256  (3 x 16 tiles)
    const float* __restrict__ b1,
    const unsigned short* __restrict__ W2p,    // K=256, N=256     (8 x 16 tiles)
    const float* __restrict__ b2,
    unsigned short* __restrict__ xplain) {     // [51200 x 256] bf16
  __shared__ __align__(64) unsigned short A1[3 * 512];
  __shared__ __align__(64) unsigned short H1[8 * 512];
  const int mt = blockIdx.x;  // 0..3199
  const int tid = threadIdx.x, lane = tid & 31, wave = tid >> 5;

  // gather teacher-forced (shifted) inputs into A fragments, K padded to 96
  for (int id = tid; id < 3 * 512; id += 256) {
    int j = id & 15, ln = (id >> 4) & 31, kt = id >> 9;
    int m = ln & 15;
    int k = kt * 32 + (j & 7) + 8 * (ln >> 4) + 16 * (j >> 3);
    int row = mt * 16 + m, s = row >> 6, bb = row & 63;
    float v0 = 0.f;
    if (s > 0 && k < N_MEL) v0 = dec_in[((size_t)(s - 1) * 64 + bb) * N_MEL + k];
    A1[id] = f2bf(v0);
  }
  __syncthreads();

  // stage 1: 80(->96) -> 256, ReLU, write into stage-2 A-fragment order
  for (int q = 0; q < 2; ++q) {
    int nt = wave * 2 + q;
    v8f acc = {};
    for (int kt = 0; kt < 3; ++kt) {
      v16bf a = *reinterpret_cast<const v16bf*>(&A1[kt * 512 + lane * 16]);
      v16bf b = *reinterpret_cast<const v16bf*>(&W1p[(kt * 16 + nt) * 512 + lane * 16]);
      acc = wmma_bf16(a, b, acc);
    }
    int nl = lane & 15, mh = 8 * (lane >> 4);
#pragma unroll
    for (int r = 0; r < 8; ++r) {
      int n = nt * 16 + nl, ml = r + mh;
      float v0 = acc[r] + b1[n];
      v0 = v0 > 0.f ? v0 : 0.f;
      int kt2 = n >> 5, kl = n & 31;
      int ln2 = ml + 16 * ((kl >> 3) & 1);
      int j2 = (kl & 7) + 8 * (kl >> 4);
      H1[kt2 * 512 + ln2 * 16 + j2] = f2bf(v0);
    }
  }
  __syncthreads();

  // stage 2: 256 -> 256, ReLU, write plain bf16 rows
  for (int q = 0; q < 2; ++q) {
    int nt = wave * 2 + q;
    v8f acc = {};
    for (int kt = 0; kt < 8; ++kt) {
      v16bf a = *reinterpret_cast<const v16bf*>(&H1[kt * 512 + lane * 16]);
      v16bf b = *reinterpret_cast<const v16bf*>(&W2p[(kt * 16 + nt) * 512 + lane * 16]);
      acc = wmma_bf16(a, b, acc);
    }
    int nl = lane & 15, mh = 8 * (lane >> 4);
#pragma unroll
    for (int r = 0; r < 8; ++r) {
      int n = nt * 16 + nl, ml = r + mh;
      float v0 = acc[r] + b2[n];
      v0 = v0 > 0.f ? v0 : 0.f;
      int row = mt * 16 + ml;
      xplain[(size_t)row * 256 + n] = f2bf(v0);
    }
  }
}

// ---------------------------------------------------------------------------
extern "C" void kernel_launch(void* const* d_in, const int* in_sizes, int n_in,
                              void* d_out, int out_size, void* d_ws, size_t ws_size,
                              hipStream_t stream) {
  (void)in_sizes; (void)n_in; (void)out_size; (void)ws_size;

  const float* memory = (const float*)d_in[0];
  const float* dec_in = (const float*)d_in[1];
  const int*   mlen   = (const int*)d_in[2];
  const float* W1 = (const float*)d_in[3];   const float* b1 = (const float*)d_in[4];
  const float* W2 = (const float*)d_in[5];   const float* b2 = (const float*)d_in[6];
  const float* Wi_a = (const float*)d_in[7]; const float* Wh_a = (const float*)d_in[8];
  const float* bi_a = (const float*)d_in[9]; const float* bh_a = (const float*)d_in[10];
  const float* Wq = (const float*)d_in[11];  const float* Wm = (const float*)d_in[12];
  const float* vv = (const float*)d_in[13];  const float* Wloc = (const float*)d_in[14];
  const float* Wld = (const float*)d_in[15];
  const float* Wi_d = (const float*)d_in[16]; const float* Wh_d = (const float*)d_in[17];
  const float* bi_d = (const float*)d_in[18]; const float* bh_d = (const float*)d_in[19];
  const float* Wp = (const float*)d_in[20];  const float* bp = (const float*)d_in[21];
  const float* Wg = (const float*)d_in[22];  const float* bg = (const float*)d_in[23];

  float* mel_out  = (float*)d_out;
  float* gate_out = mel_out + (size_t)B_SZ * T_DEC * N_MEL;
  float* align_out = gate_out + (size_t)B_SZ * T_DEC;

  // ---- workspace carve-out (~116 MB; LSTM weights+memory stay L2-resident)
  char* ws = (char*)d_ws;
  size_t off = 0;
  auto carve = [&](size_t bytes) -> char* {
    char* p = ws + off;
    off = (off + bytes + 255) & ~(size_t)255;
    return p;
  };
  unsigned short* WaP = (unsigned short*)carve((size_t)KT_ATTN * NT_GATES * 512 * 2); // 14.7 MB
  unsigned short* WdP = (unsigned short*)carve((size_t)KT_DEC * NT_GATES * 512 * 2);  // 21.0 MB
  unsigned short* W1P = (unsigned short*)carve((size_t)3 * 16 * 512 * 2);
  unsigned short* W2P = (unsigned short*)carve((size_t)8 * 16 * 512 * 2);
  float* bias_a = (float*)carve(4096 * 4);
  float* bias_d = (float*)carve(4096 * 4);
  unsigned short* xplain = (unsigned short*)carve((size_t)T_DEC * 64 * 256 * 2);      // 26.2 MB
  float* pmem = (float*)carve((size_t)64 * 512 * 128 * 4);                            // 16.8 MB
  unsigned short* membf = (unsigned short*)carve((size_t)64 * 512 * 512 * 2);         // 33.6 MB
  float* gates = (float*)carve((size_t)64 * 4096 * 4);
  unsigned short* Aattn = (unsigned short*)carve((size_t)KT_ATTN * 4 * 512 * 2);
  unsigned short* Adec  = (unsigned short*)carve((size_t)KT_DEC * 4 * 512 * 2);
  float* ah = (float*)carve(65536 * 4);
  float* ac = (float*)carve(65536 * 4);
  float* dh = (float*)carve(65536 * 4);
  float* dc = (float*)carve(65536 * 4);
  float* aw  = (float*)carve(32768 * 4);
  float* awc = (float*)carve(32768 * 4);
  float* ctx = (float*)carve(32768 * 4);

  // ---- one-time prep (per call; deterministic) ----
  {
    long tot = (long)KT_ATTN * NT_GATES * 512;
    pack_bfrag<<<(tot + 255) / 256, 256, 0, stream>>>(WaP, Wi_a, 768, 768, Wh_a, 1024, K_ATTN, NT_GATES);
  }
  {
    long tot = (long)KT_DEC * NT_GATES * 512;
    pack_bfrag<<<(tot + 255) / 256, 256, 0, stream>>>(WdP, Wi_d, 1536, 1536, Wh_d, 1024, K_DEC, NT_GATES);
  }
  pack_bfrag<<<(3 * 16 * 512 + 255) / 256, 256, 0, stream>>>(W1P, W1, 80, 80, nullptr, 0, 96, 16);
  pack_bfrag<<<(8 * 16 * 512 + 255) / 256, 256, 0, stream>>>(W2P, W2, 256, 256, nullptr, 0, 256, 16);
  init_kernel<<<256, 256, 0, stream>>>(ah, ac, dh, dc, aw, awc, ctx,
                                       bi_a, bh_a, bias_a, bi_d, bh_d, bias_d);
  pmem_kernel<<<B_SZ * T_ENC, 256, 0, stream>>>(memory, Wm, pmem, membf);
  prenet_kernel<<<(T_DEC * B_SZ) / 16, 256, 0, stream>>>(dec_in, W1P, b1, W2P, b2, xplain);

  // ---- recurrent loop: weights stream from L2 every step ----
  for (int s = 0; s < T_DEC; ++s) {
    // attention LSTM: A = concat(x_t[256], ctx[512], ah[1024])
    packA<<<(KT_ATTN * 4 * 512) / 256, 256, 0, stream>>>(
        Aattn, KT_ATTN, xplain + (size_t)s * 64 * 256, 256, ctx, 512, ah, 1024, ah, 0);
    lstm_gemm<<<128, 256, 0, stream>>>(Aattn, WaP, bias_a, gates, KT_ATTN);
    lstm_point<<<256, 256, 0, stream>>>(gates, ah, ac);

    attn_kernel<<<B_SZ, 256, 0, stream>>>(ah, Wq, pmem, aw, awc, Wloc, Wld, vv,
                                          membf, mlen, ctx, align_out, s);

    // decoder LSTM: A = concat(ah[1024], ctx[512], dh[1024])
    packA<<<(KT_DEC * 4 * 512) / 256, 256, 0, stream>>>(
        Adec, KT_DEC, nullptr, 0, ah, 1024, ctx, 512, dh, 1024);
    lstm_gemm<<<128, 256, 0, stream>>>(Adec, WdP, bias_d, gates, KT_DEC);
    lstm_point<<<256, 256, 0, stream>>>(gates, dh, dc);

    proj_kernel<<<B_SZ, 128, 0, stream>>>(dh, ctx, Wp, bp, Wg, bg, mel_out, gate_out, s);
  }
}
